// LSTM_40235253629059
// MI455X (gfx1250) — matrix operand order repaired and checked by reading, not compiled, not run on hardware
//
#include <hip/hip_runtime.h>
#include <stdint.h>

// =====================================================================
// 2-layer LSTM (B=16, T=2048, H=512, I=32) for MI455X (gfx1250).
//
// The recurrence is latency-bound (33.5 MFLOP/step, 2048 serial steps).
// Persistent kernel: 16 workgroups (1/WGP), each holds a 128-row slice of
// [W_hh | W_ih] in LDS (layer1: 256KB -- needs CDNA5's 320KB LDS/WGP).
// Per step, each of 8 waves runs Ktot/32 v_wmma_f32_16x16x32_bf16 (fp32
// accumulate, bias preloaded in C). h ping-pongs through a 16KB L2-resident
// buffer; steps sync via a monotonic device-scope atomic barrier.
// Staging uses GLOBAL_LOAD_ASYNC_TO_LDS_B128 (+ s_wait_asynccnt).
// Activations use native v_exp/v_rcp (TRANS32 ops co-execute with WMMA).
// =====================================================================

typedef __attribute__((ext_vector_type(16))) __bf16 v16bf;
typedef __attribute__((ext_vector_type(8)))  float  v8f;
typedef __attribute__((ext_vector_type(4)))  int    v4i;

// address-space-qualified int4 pointee types for the async-LDS builtin
typedef __attribute__((address_space(1))) v4i gv4i;   // global
typedef __attribute__((address_space(3))) v4i lv4i;   // LDS

union FragU { uint4 q[2]; v16bf v; };

#define LSTM_B   16
#define LSTM_T   2048
#define LSTM_H   512
#define LSTM_NWG 16

// ---------------- async global->LDS copy (CDNA5), guarded ----------------
#if defined(__AMDGCN__) && __has_builtin(__builtin_amdgcn_global_load_async_to_lds_b128)
#define HAVE_ASYNC_LDS 1
#endif

__device__ __forceinline__ void cp16_g2l(void* lds, const void* g) {
#ifdef HAVE_ASYNC_LDS
  __builtin_amdgcn_global_load_async_to_lds_b128(
      (gv4i*)(uintptr_t)g,   // global src: AS1 int4*
      (lv4i*)lds,            // LDS dst: AS3 int4* (addrspacecast)
      0, 0);
#else
  *(uint4*)lds = *(const uint4*)g;
#endif
}

__device__ __forceinline__ void wait_cp_g2l() {
#ifdef HAVE_ASYNC_LDS
#if __has_builtin(__builtin_amdgcn_s_wait_asynccnt)
  __builtin_amdgcn_s_wait_asynccnt(0);
#else
  asm volatile("s_wait_asynccnt 0" ::: "memory");
#endif
#endif
}

// ---------------- fast activations (native TRANS32 ops) ----------------
__device__ __forceinline__ float fast_sigmoid(float x) {
#if defined(__AMDGCN__) && __has_builtin(__builtin_amdgcn_exp2f) && \
    __has_builtin(__builtin_amdgcn_rcpf)
  // sigmoid(x) = 1 / (1 + 2^(-x*log2(e))): v_exp_f32 + v_rcp_f32
  float e = __builtin_amdgcn_exp2f(x * -1.442695041f);
  return __builtin_amdgcn_rcpf(1.0f + e);
#else
  return 1.0f / (1.0f + __expf(-x));
#endif
}

__device__ __forceinline__ float fast_tanh(float x) {
#if defined(__AMDGCN__) && __has_builtin(__builtin_amdgcn_tanhf)
  return __builtin_amdgcn_tanhf(x);           // native v_tanh_f32 if present
#else
  return 2.0f * fast_sigmoid(2.0f * x) - 1.0f;
#endif
}

// ---------------- bf16 helpers ----------------
__device__ __forceinline__ unsigned int f2bf_raw(float x) {
  unsigned int u = __float_as_uint(x);
  u += 0x7fffu + ((u >> 16) & 1u);   // round-to-nearest-even
  return u >> 16;
}

// ---------------- WMMA fragment loads (ISA 7.12.2 layouts) ----------------
// A: 16x32 bf16, row-major [M][K] source. Lane m=lane&15; lanes>=16 take
// K-base +8; per lane: 8 elems at (m, k0+kb) and 8 at (m, k0+kb+16).
__device__ __forceinline__ v16bf load_frag_a(const unsigned short* base, int lda,
                                             int k0, int lane) {
  int m  = lane & 15;
  int kb = (lane >> 4) ? 8 : 0;
  FragU f;
  f.q[0] = *(const uint4*)(base + m * lda + k0 + kb);
  f.q[1] = *(const uint4*)(base + m * lda + k0 + kb + 16);
  return f.v;
}

// B: 32x16 bf16 (K x N), source stored row-major [N][K] (i.e. w[n][k]).
// Lane n=lane&15; lanes>=16 take K-base +8; 8 elems at k0+kb and 8 at k0+kb+16.
__device__ __forceinline__ v16bf load_frag_b(const unsigned short* base, int ldb,
                                             int k0, int lane) {
  int n  = lane & 15;
  int kb = (lane >> 4) ? 8 : 0;
  FragU f;
  f.q[0] = *(const uint4*)(base + n * ldb + k0 + kb);
  f.q[1] = *(const uint4*)(base + n * ldb + k0 + kb + 16);
  return f.v;
}

__device__ __forceinline__ v8f wmma_bf16(v16bf a, v16bf b, v8f c) {
  return __builtin_amdgcn_wmma_f32_16x16x32_bf16(false, a, false, b,
                                                 (short)0, c, false, false);
}

// ---------------- fp32 -> bf16 conversion ----------------
__global__ void cvt_f32_bf16(const float* __restrict__ in,
                             unsigned short* __restrict__ out, int n) {
  int i = blockIdx.x * blockDim.x + threadIdx.x;
  if (i < n) out[i] = (unsigned short)f2bf_raw(in[i]);
}

// ---------------- per-launch state init (determinism) ----------------
__global__ void init_state_kernel(unsigned short* hbuf, unsigned int* bar) {
  int i = blockIdx.x * blockDim.x + threadIdx.x;
  if (i < 2 * LSTM_B * LSTM_H) hbuf[i] = 0;
  if (i == 0) *bar = 0u;
}

// ---------------- persistent LSTM layer kernel ----------------
// Grid = 16 WGs x 256 threads. WG b owns H-columns [32b, 32b+32) -> the
// 128 gate rows {g*512 + 32b + lc : g in 0..3, lc in 0..31}.
// Combined step GEMM: gates[16,128] = [h_{t-1} | x_t][16,Ktot] x Wslice^T,
// Ktot = 512 + KIN (layer0: 544, layer1: 1024). KIN is a template param so
// every loop has compile-time bounds (enables unroll + load pipelining).
template <int KIN>
__global__ void __launch_bounds__(256) lstm_layer_kernel(
    const unsigned short* __restrict__ xin,   // [B,T,KIN] bf16
    const unsigned short* __restrict__ whh,   // [4H,512] bf16
    const unsigned short* __restrict__ wih,   // [4H,KIN] bf16
    const float* __restrict__ b_ih,
    const float* __restrict__ b_hh,
    unsigned short* __restrict__ hseq,        // [B,T,512] bf16 (layer output)
    float* __restrict__ hn,                   // [B,512] f32 final h
    float* __restrict__ cn,                   // [B,512] f32 final c
    unsigned short* __restrict__ hbuf,        // [2][B][512] bf16 ping-pong
    unsigned int* __restrict__ bar) {         // monotonic step barrier
  constexpr int H = LSTM_H, T = LSTM_T;
  constexpr int Ktot = 512 + KIN;
  constexpr int kc   = Ktot >> 3;             // 16B chunks per row
  constexpr int hc   = (16 * Ktot) >> 3;

  extern __shared__ char smem[];
  unsigned short* w_lds = (unsigned short*)smem;        // [128][Ktot]
  unsigned short* h_lds = w_lds + 128 * Ktot;           // [16][Ktot]
  float* g_lds = (float*)(h_lds + 16 * Ktot);           // [4][16][32]
  float* c_lds = g_lds + 4 * 16 * 32;                   // [16][32]

  const int tid  = threadIdx.x;
  const int lane = tid & 31;
  const int wave = tid >> 5;            // 0..7 : n-tile within the 128-slice
  const int j0   = blockIdx.x * 32;     // H-column base

  // zero persistent cell state
  for (int e = tid; e < 16 * 32; e += 256) c_lds[e] = 0.f;

  // ---- stage weight slice into LDS (once; resident for all T steps) ----
  for (int q = tid; q < 128 * kc; q += 256) {
    int r  = q / kc;
    int cc = (q - r * kc) * 8;
    int gate = r >> 5, lc = r & 31;
    int gr = gate * H + j0 + lc;        // global gate row (i,f,g,o order)
    const unsigned short* src =
        (cc < 512) ? (whh + gr * 512 + cc) : (wih + gr * KIN + (cc - 512));
    cp16_g2l(w_lds + r * Ktot + cc, src);
  }

  // this wave's output column -> gate type + bias (loop-invariant)
  const int nt     = wave;
  const int gate_w = nt >> 1;
  const int lcol   = ((nt & 1) << 4) + (lane & 15);
  const int gr_w   = gate_w * H + j0 + lcol;
  const float bias = b_ih[gr_w] + b_hh[gr_w];
  // branch-free gate nonlinearity: tanh(x) = 2*sigmoid(2x) - 1
  const float act_scl = (gate_w == 2) ? 2.0f : 1.0f;
  const float act_b   = (gate_w == 2) ? -1.0f : 0.0f;

  wait_cp_g2l();
  __syncthreads();

  for (int t = 0; t < T; ++t) {
    // ---- stage A = [h_{t-1} | x_t] into LDS ----
    const unsigned short* hb = hbuf + (t & 1) * (LSTM_B * H);
    for (int q = tid; q < hc; q += 256) {
      int row = q / kc;
      int cc  = (q - row * kc) * 8;
      const unsigned short* src =
          (cc < 512) ? (hb + row * H + cc)
                     : (xin + (row * T + t) * KIN + (cc - 512));
      cp16_g2l(h_lds + row * Ktot + cc, src);
    }
    wait_cp_g2l();
    __syncthreads();

    // ---- gates[16,16] for this wave's tile: WMMA over Ktot ----
    v8f acc;
#pragma unroll
    for (int r = 0; r < 8; ++r) acc[r] = bias;
#pragma unroll 4
    for (int ks = 0; ks < Ktot; ks += 32) {
      v16bf a = load_frag_a(h_lds, Ktot, ks, lane);
      v16bf b = load_frag_b(w_lds + nt * 16 * Ktot, Ktot, ks, lane);
      acc = wmma_bf16(a, b, acc);
    }

    // ---- activations (one v_exp + v_rcp each), scatter to gate scratch ----
#pragma unroll
    for (int r = 0; r < 8; ++r) {
      int m   = r + ((lane >> 4) << 3);      // batch row
      float g = fast_sigmoid(act_scl * acc[r]) * act_scl + act_b;
      g_lds[(gate_w * 16 + m) * 32 + lcol] = g;
    }
    __syncthreads();

    // ---- elementwise cell/h update: 512 elems, 2 adjacent cols/thread ----
    unsigned short* hw = hbuf + ((t + 1) & 1) * (LSTM_B * H);
    {
      int e2 = tid * 2;                 // 0..510
      int m = e2 >> 5, lc = e2 & 31;
      float2 gi = *(const float2*)&g_lds[(0 * 16 + m) * 32 + lc];
      float2 gf = *(const float2*)&g_lds[(1 * 16 + m) * 32 + lc];
      float2 gg = *(const float2*)&g_lds[(2 * 16 + m) * 32 + lc];
      float2 go = *(const float2*)&g_lds[(3 * 16 + m) * 32 + lc];
      float2 cprev = *(const float2*)&c_lds[e2];
      float c0 = gf.x * cprev.x + gi.x * gg.x;
      float c1 = gf.y * cprev.y + gi.y * gg.y;
      *(float2*)&c_lds[e2] = make_float2(c0, c1);
      float h0 = go.x * fast_tanh(c0);
      float h1 = go.y * fast_tanh(c1);
      unsigned int hp = f2bf_raw(h0) | (f2bf_raw(h1) << 16);
      *(unsigned int*)(hw + m * H + j0 + lc)             = hp;
      *(unsigned int*)(hseq + (m * T + t) * H + j0 + lc) = hp;
      if (t == T - 1) {
        *(float2*)&hn[m * H + j0 + lc] = make_float2(h0, h1);
        *(float2*)&cn[m * H + j0 + lc] = make_float2(c0, c1);
      }
    }

    // ---- device-wide step barrier (16 resident WGs, monotonic count) ----
    __threadfence();
    __syncthreads();
    if (tid == 0) {
      __hip_atomic_fetch_add(bar, 1u, __ATOMIC_ACQ_REL, __HIP_MEMORY_SCOPE_AGENT);
      unsigned int target = (unsigned int)LSTM_NWG * (unsigned int)(t + 1);
      while (__hip_atomic_load(bar, __ATOMIC_ACQUIRE, __HIP_MEMORY_SCOPE_AGENT) < target)
        __builtin_amdgcn_s_sleep(1);
    }
    __syncthreads();
  }
}

// ---------------- output projection: y = h2 @ w_out^T + b_out ----------------
// M = B*T = 32768 rows, N = 32, K = 512. One wave per 16-row strip; the two
// 16-col n-tiles share the A fragments.
__global__ void __launch_bounds__(256) proj_kernel(
    const unsigned short* __restrict__ h2,     // [32768,512] bf16
    const unsigned short* __restrict__ wout,   // [32,512] bf16
    const float* __restrict__ bout,            // [32]
    float* __restrict__ y) {                   // [32768,32]
  int gwave = (blockIdx.x * blockDim.x + threadIdx.x) >> 5;  // 0..2047
  int lane  = threadIdx.x & 31;
  int m0    = gwave * 16;
  v8f acc0 = {}, acc1 = {};
#pragma unroll 2
  for (int ks = 0; ks < 512; ks += 32) {
    v16bf a  = load_frag_a(h2 + m0 * 512, 512, ks, lane);
    v16bf b0 = load_frag_b(wout,            512, ks, lane);
    v16bf b1 = load_frag_b(wout + 16 * 512, 512, ks, lane);
    acc0 = wmma_bf16(a, b0, acc0);
    acc1 = wmma_bf16(a, b1, acc1);
  }
  int n  = lane & 15;
  int mh = (lane >> 4) << 3;
#pragma unroll
  for (int r = 0; r < 8; ++r) {
    int row = m0 + r + mh;
    y[row * 32 + n]      = acc0[r] + bout[n];
    y[row * 32 + 16 + n] = acc1[r] + bout[16 + n];
  }
}

// =====================================================================
// host launcher
// =====================================================================
extern "C" void kernel_launch(void* const* d_in, const int* in_sizes, int n_in,
                              void* d_out, int out_size, void* d_ws, size_t ws_size,
                              hipStream_t stream) {
  (void)in_sizes; (void)n_in; (void)out_size; (void)ws_size;
  const float* x     = (const float*)d_in[0];
  const float* w_ih0 = (const float*)d_in[1];
  const float* w_hh0 = (const float*)d_in[2];
  const float* b_ih0 = (const float*)d_in[3];
  const float* b_hh0 = (const float*)d_in[4];
  const float* w_ih1 = (const float*)d_in[5];
  const float* w_hh1 = (const float*)d_in[6];
  const float* b_ih1 = (const float*)d_in[7];
  const float* b_hh1 = (const float*)d_in[8];
  const float* w_out = (const float*)d_in[9];
  const float* b_out = (const float*)d_in[10];
  float* out = (float*)d_out;

  // workspace carve-up (256B aligned)
  char* ws = (char*)d_ws;
  size_t off = 0;
  auto carve = [&](size_t bytes) {
    char* p = ws + off;
    off += (bytes + 255) & ~(size_t)255;
    return p;
  };
  unsigned short* x_bf    = (unsigned short*)carve((size_t)LSTM_B * LSTM_T * 32 * 2);
  unsigned short* wih0_bf = (unsigned short*)carve((size_t)2048 * 32 * 2);
  unsigned short* whh0_bf = (unsigned short*)carve((size_t)2048 * 512 * 2);
  unsigned short* wih1_bf = (unsigned short*)carve((size_t)2048 * 512 * 2);
  unsigned short* whh1_bf = (unsigned short*)carve((size_t)2048 * 512 * 2);
  unsigned short* wout_bf = (unsigned short*)carve((size_t)32 * 512 * 2);
  unsigned short* h1_bf   = (unsigned short*)carve((size_t)LSTM_B * LSTM_T * 512 * 2);
  unsigned short* h2_bf   = (unsigned short*)carve((size_t)LSTM_B * LSTM_T * 512 * 2);
  unsigned short* hbuf    = (unsigned short*)carve((size_t)2 * LSTM_B * 512 * 2);
  unsigned int*   bar     = (unsigned int*)carve(256);

  // d_out layout: y[16,2048,32] | h_n[2,16,512] | c_n[2,16,512]
  float* y   = out;
  float* hn0 = out + 1048576;
  float* hn1 = hn0 + 8192;
  float* cn0 = out + 1048576 + 16384;
  float* cn1 = cn0 + 8192;

  const int smem0 = 144 * (512 + 32)  * 2 + 10240;  // 166,912 B
  const int smem1 = 144 * (512 + 512) * 2 + 10240;  // 305,152 B

  // allow >64KB dynamic LDS for the persistent layer kernels
  (void)hipFuncSetAttribute((const void*)lstm_layer_kernel<32>,
                            hipFuncAttributeMaxDynamicSharedMemorySize, smem0);
  (void)hipFuncSetAttribute((const void*)lstm_layer_kernel<512>,
                            hipFuncAttributeMaxDynamicSharedMemorySize, smem1);

  // fp32 -> bf16 conversions
  auto cvt = [&](const float* in, unsigned short* o, int n) {
    cvt_f32_bf16<<<(n + 255) / 256, 256, 0, stream>>>(in, o, n);
  };
  cvt(x,     x_bf,    LSTM_B * LSTM_T * 32);
  cvt(w_ih0, wih0_bf, 2048 * 32);
  cvt(w_hh0, whh0_bf, 2048 * 512);
  cvt(w_ih1, wih1_bf, 2048 * 512);
  cvt(w_hh1, whh1_bf, 2048 * 512);
  cvt(w_out, wout_bf, 32 * 512);

  // ---- layer 0 ----
  init_state_kernel<<<64, 256, 0, stream>>>(hbuf, bar);
  lstm_layer_kernel<32><<<LSTM_NWG, 256, smem0, stream>>>(
      x_bf, whh0_bf, wih0_bf, b_ih0, b_hh0,
      h1_bf, hn0, cn0, hbuf, bar);

  // ---- layer 1 ----
  init_state_kernel<<<64, 256, 0, stream>>>(hbuf, bar);
  lstm_layer_kernel<512><<<LSTM_NWG, 256, smem1, stream>>>(
      h1_bf, whh1_bf, wih1_bf, b_ih1, b_hh1,
      h2_bf, hn1, cn1, hbuf, bar);

  // ---- output projection ----
  proj_kernel<<<256, 256, 0, stream>>>(h2_bf, wout_bf, b_out, y);
}
